// LASAGESConv_22926535426631
// MI455X (gfx1250) — matrix operand order repaired and checked by reading, not compiled, not run on hardware
//
#include <hip/hip_runtime.h>
#include <hip/hip_bf16.h>

typedef __attribute__((ext_vector_type(16))) _Float16 v16h;
typedef __attribute__((ext_vector_type(8)))  float    v8f;

#define TW 136   // padded LDS tile width (128 + 8) to dodge bank conflicts

// ---------------------------------------------------------------------------
// Fragment builders (CDNA5 wave32 WMMA layouts, CDNA5 ISA §7.12.2)
// A 16x32 f16: lanes 0-15 -> rows M, K = koff..koff+7 in v[0..7],
//              K = koff+16.. in v[8..15]; lanes 16-31 get koff+8.
// B 32x16 f16: lanes 0-15 -> cols N, K=0..7,16..23; lanes 16-31 -> K=8..15,24..31.
// C/D f32:     VGPR r -> row r (lanes 0-15) / row r+8 (lanes 16-31), col=lane&15.
// ---------------------------------------------------------------------------

__device__ __forceinline__ v16h load_a_lds(const _Float16 T[][TW], int kb, int lane) {
  const int m    = lane & 15;
  const int koff = kb * 32 + ((lane & 16) ? 8 : 0);
  v16h a;
#pragma unroll
  for (int v = 0; v < 4; ++v) {
    a[2 * v]         = T[m][koff + 2 * v];
    a[2 * v + 1]     = T[m][koff + 2 * v + 1];
    a[8 + 2 * v]     = T[m][koff + 16 + 2 * v];
    a[8 + 2 * v + 1] = T[m][koff + 16 + 2 * v + 1];
  }
  return a;
}

__device__ __forceinline__ v16h load_a_mul(const _Float16 S[][TW], const _Float16 G[][TW],
                                           int kb, int lane) {
  const int m    = lane & 15;
  const int koff = kb * 32 + ((lane & 16) ? 8 : 0);
  v16h a;
#pragma unroll
  for (int v = 0; v < 4; ++v) {
    int k0 = koff + 2 * v;
    int k1 = koff + 16 + 2 * v;
    a[2 * v]         = (_Float16)(S[m][k0]     * G[m][k0]);
    a[2 * v + 1]     = (_Float16)(S[m][k0 + 1] * G[m][k0 + 1]);
    a[8 + 2 * v]     = (_Float16)(S[m][k1]     * G[m][k1]);
    a[8 + 2 * v + 1] = (_Float16)(S[m][k1 + 1] * G[m][k1 + 1]);
  }
  return a;
}

// B fragment from pre-converted f16 row-major [out][in] weight (B = W^T).
// Two 8-half chunks per lane (K = koff..koff+7, koff+16..koff+23; koff += 8 for lanes 16-31),
// 16-B aligned -> two global_load_b128, no cvt.
__device__ __forceinline__ v16h load_b_h(const _Float16* __restrict__ W, int nb, int kb, int lane) {
  typedef __attribute__((ext_vector_type(8))) _Float16 v8h_t;
  const int n    = nb + (lane & 15);
  const int koff = kb * 32 + ((lane & 16) ? 8 : 0);
  const v8h_t lo = *(const v8h_t*)(W + n * 128 + koff);
  const v8h_t hi = *(const v8h_t*)(W + n * 128 + koff + 16);
  return __builtin_shufflevector(lo, hi, 0, 1, 2, 3, 4, 5, 6, 7, 8, 9, 10, 11, 12, 13, 14, 15);
}

__device__ __forceinline__ v8f wmma_acc(v16h a, v16h b, v8f c) {
  return __builtin_amdgcn_wmma_f32_16x16x32_f16(false, a, false, b, (short)0, c, false, false);
}

// [16,128] x [128,128]^T column-tile GEMM, A fragments pre-built (h reused 6x)
__device__ __forceinline__ v8f mm_h(const v16h* ah, const _Float16* __restrict__ W,
                                    int nb, int lane) {
  v8f acc = {};
#pragma unroll
  for (int kb = 0; kb < 4; ++kb) acc = wmma_acc(ah[kb], load_b_h(W, nb, kb, lane), acc);
  return acc;
}

__device__ __forceinline__ v8f mm_lds(const _Float16 A[][TW], const _Float16* __restrict__ W,
                                      int nb, int lane) {
  v8f acc = {};
#pragma unroll
  for (int kb = 0; kb < 4; ++kb)
    acc = wmma_acc(load_a_lds(A, kb, lane), load_b_h(W, nb, kb, lane), acc);
  return acc;
}

__device__ __forceinline__ v8f mm_mul(const _Float16 S[][TW], const _Float16 G[][TW],
                                      const _Float16* __restrict__ W, int nb, int lane) {
  v8f acc = {};
#pragma unroll
  for (int kb = 0; kb < 4; ++kb)
    acc = wmma_acc(load_a_mul(S, G, kb, lane), load_b_h(W, nb, kb, lane), acc);
  return acc;
}

__device__ __forceinline__ void store_tile(v8f acc, const float* __restrict__ bias,
                                           _Float16 T[][TW], int col, int rbase, bool relu) {
  const float b = bias[col];
#pragma unroll
  for (int r = 0; r < 8; ++r) {
    float v = acc[r] + b;
    if (relu) v = fmaxf(v, 0.f);
    T[rbase + r][col] = (_Float16)v;
  }
}

__device__ __forceinline__ void load8(const float* __restrict__ g, _Float16* t) {
  const float4 v0 = ((const float4*)g)[0];
  const float4 v1 = ((const float4*)g)[1];
  t[0] = (_Float16)v0.x; t[1] = (_Float16)v0.y; t[2] = (_Float16)v0.z; t[3] = (_Float16)v0.w;
  t[4] = (_Float16)v1.x; t[5] = (_Float16)v1.y; t[6] = (_Float16)v1.z; t[7] = (_Float16)v1.w;
}

// ---------------------------------------------------------------------------
// Phase 0a: zero the three segment accumulators in workspace
// ---------------------------------------------------------------------------
__global__ void lasage_zero_kernel(float* __restrict__ p, size_t n) {
  size_t i = (size_t)blockIdx.x * blockDim.x + threadIdx.x;
  if (i < n) p[i] = 0.f;
}

// ---------------------------------------------------------------------------
// Phase 0b: one-shot f32 -> f16 conversion of the ten 128x128 weights into a
// 320 KB L2-resident pool (removes ~450 v_cvt per wave from the dense loop
// and halves the weight-fetch bytes).
// ---------------------------------------------------------------------------
struct WPtrs { const float* w[10]; };

__global__ void lasage_wcvt_kernel(WPtrs ws, _Float16* __restrict__ dst) {
  const int i = blockIdx.x * blockDim.x + threadIdx.x;   // 10 * 16384 total
  const int m = i >> 14;
  const int e = i & 16383;
  dst[i] = (_Float16)ws.w[m][e];
}

// f16 pool layout (each 128*128 = 16384 halves)
#define W_FRW1  0
#define W_FRW2  1
#define W_FRT1  2
#define W_FRT2  3
#define W_BEW1  4
#define W_BEW2  5
#define W_BET1  6
#define W_BET2  7
#define W_BAL1  8
#define W_SELF  9

// ---------------------------------------------------------------------------
// Phase 1: per-label segment-sum scatter.  32 lanes per edge, float4 per lane.
// Segment buffers (153.6 MB) fit the 192 MB L2, so atomics resolve near-chip.
// ---------------------------------------------------------------------------
__global__ void lasage_scatter_kernel(const float* __restrict__ feat,
                                      const int* __restrict__ src,
                                      const int* __restrict__ dst,
                                      const int* __restrict__ labels,
                                      float* __restrict__ seg_fr,
                                      float* __restrict__ seg_be,
                                      float* __restrict__ seg_unk,
                                      int E) {
  const long long t = (long long)blockIdx.x * blockDim.x + threadIdx.x;
  const long long e = t >> 5;
  if (e >= E) return;
  const int q = (int)(t & 31) * 4;
  const int s = src[e];
  const int d = dst[e];
  const int l = labels[s];
  const float4 v = *(const float4*)(feat + (size_t)s * 128 + q);
  float* seg = (l == 1) ? seg_fr : ((l == 0) ? seg_be : seg_unk);
  float* o = seg + (size_t)d * 128 + q;
  atomicAdd(o + 0, v.x);
  atomicAdd(o + 1, v.y);
  atomicAdd(o + 2, v.z);
  atomicAdd(o + 3, v.w);
}

// ---------------------------------------------------------------------------
// Phase 2: fused dense pipeline. Block = 16 nodes, 8 waves; wave owns a
// 16-wide output column tile. 14 GEMMs/node-tile -> 56 v_wmma per wave.
// ---------------------------------------------------------------------------
struct DenseArgs {
  const float    *feat, *sfr, *sbe, *sunk;
  const _Float16 *wpool;                         // 10 x 16384 f16 matrices
  const float    *frb1, *frT1b, *frb2, *frT2b;
  const float    *beb1, *beT1b, *beb2, *beT2b;
  const float    *balb1, *balW2, *balb2;
  const float    *selfb;
  float*          out;
};

__global__ __launch_bounds__(256, 2)
void lasage_dense_kernel(DenseArgs A, int N) {
  __shared__ _Float16 Ah [16][TW];
  __shared__ _Float16 Sfr[16][TW], Sbe[16][TW], Sun[16][TW];
  __shared__ _Float16 G1f[16][TW], G2f[16][TW], G1b[16][TW], G2b[16][TW];
  __shared__ _Float16 Xb [16][TW], BalH[16][TW];
  __shared__ float    balv[16];

  const int tid   = threadIdx.x;
  const int lane  = tid & 31;
  const int wid   = tid >> 5;
  const int nb    = wid * 16;
  const int node0 = blockIdx.x * 16;

  const _Float16* __restrict__ WP = A.wpool;

  // cooperative f32->f16 tile loads (8 contiguous elems per thread per tile)
  {
    const int idx = tid * 8;
    const int row = idx >> 7, col = idx & 127;
    int rg = node0 + row; if (rg > N - 1) rg = N - 1;   // clamp (N % 16 == 0 anyway)
    const size_t off = (size_t)rg * 128 + col;
    load8(A.feat + off, &Ah [row][col]);
    load8(A.sfr  + off, &Sfr[row][col]);
    load8(A.sbe  + off, &Sbe[row][col]);
    load8(A.sunk + off, &Sun[row][col]);
  }
  __syncthreads();

  // h fragments, reused by 6 GEMMs
  v16h ah[4];
#pragma unroll
  for (int kb = 0; kb < 4; ++kb) ah[kb] = load_a_lds(Ah, kb, lane);

  const int col   = nb + (lane & 15);
  const int rbase = (lane & 16) ? 8 : 0;

  // gate GEMMs (shared between neigh_* and unk_*), balance hidden, self path
  store_tile(mm_h(ah, WP + W_FRT1 * 16384, nb, lane), A.frT1b, G1f,  col, rbase, false);
  store_tile(mm_h(ah, WP + W_FRT2 * 16384, nb, lane), A.frT2b, G2f,  col, rbase, false);
  store_tile(mm_h(ah, WP + W_BET1 * 16384, nb, lane), A.beT1b, G1b,  col, rbase, false);
  store_tile(mm_h(ah, WP + W_BET2 * 16384, nb, lane), A.beT2b, G2b,  col, rbase, false);
  store_tile(mm_h(ah, WP + W_BAL1 * 16384, nb, lane), A.balb1, BalH, col, rbase, true);
  v8f acc_out = mm_h(ah, WP + W_SELF * 16384, nb, lane);
  __syncthreads();

  // balance = sigmoid(relu(h W1 + b1) w2 + b2), one thread per row
  if (tid < 16) {
    float s = A.balb2[0];
    for (int k = 0; k < 128; ++k) s += (float)BalH[tid][k] * A.balW2[k];
    balv[tid] = 1.f / (1.f + __expf(-s));
  }
  __syncthreads();

  // four gated-MLP branches: fr, be, unk*bal(fr), unk*(1-bal)(be)
  _Float16 (*Sarr[4])[TW]  = {Sfr, Sbe, Sun, Sun};
  _Float16 (*G1arr[4])[TW] = {G1f, G1b, G1f, G1b};
  _Float16 (*G2arr[4])[TW] = {G2f, G2b, G2f, G2b};
  const _Float16* W1a[4] = {WP + W_FRW1 * 16384, WP + W_BEW1 * 16384,
                            WP + W_FRW1 * 16384, WP + W_BEW1 * 16384};
  const _Float16* W2a[4] = {WP + W_FRW2 * 16384, WP + W_BEW2 * 16384,
                            WP + W_FRW2 * 16384, WP + W_BEW2 * 16384};
  const float* b1a[4] = {A.frb1, A.beb1, A.frb1, A.beb1};
  const float* b2a[4] = {A.frb2, A.beb2, A.frb2, A.beb2};

#pragma unroll
  for (int br = 0; br < 4; ++br) {
    v8f x = mm_mul(Sarr[br], G1arr[br], W1a[br], nb, lane);   // (s*g1) @ W1^T
    const float b1 = b1a[br][col];
    __syncthreads();                       // previous branch's Xb fully consumed
#pragma unroll
    for (int r = 0; r < 8; ++r)
      Xb[rbase + r][col] = (_Float16)((x[r] + b1) * (float)G2arr[br][rbase + r][col]);
    __syncthreads();
    v8f y = mm_lds(Xb, W2a[br], nb, lane); // (x*g2) @ W2^T
    const float b2 = b2a[br][col];
#pragma unroll
    for (int r = 0; r < 8; ++r) {
      const int row = rbase + r;
      const float wf = (br == 0 || br == 1) ? 1.f
                     : (br == 2 ? balv[row] : (1.f - balv[row]));
      acc_out[r] += wf * (y[r] + b2);
    }
  }

  // epilogue: + self bias, ReLU, f32 out
  const float sb = A.selfb[col];
#pragma unroll
  for (int r = 0; r < 8; ++r) {
    const int row = node0 + rbase + r;
    if (row < N) A.out[(size_t)row * 128 + col] = fmaxf(acc_out[r] + sb, 0.f);
  }
}

// ---------------------------------------------------------------------------
extern "C" void kernel_launch(void* const* d_in, const int* in_sizes, int n_in,
                              void* d_out, int out_size, void* d_ws, size_t ws_size,
                              hipStream_t stream) {
  const float* feat   = (const float*)d_in[0];
  const int*   src    = (const int*)  d_in[1];
  const int*   dst    = (const int*)  d_in[2];
  const int*   labels = (const int*)  d_in[3];

  const int E = in_sizes[1];
  const int N = in_sizes[3];

  float* sfr  = (float*)d_ws;
  float* sbe  = sfr  + (size_t)N * 128;
  float* sunk = sbe  + (size_t)N * 128;
  _Float16* wpool = (_Float16*)(sunk + (size_t)N * 128);   // 10*16384 halves

  // phase 0a: clear accumulators
  {
    size_t zn = (size_t)3 * N * 128;
    lasage_zero_kernel<<<(unsigned)((zn + 255) / 256), 256, 0, stream>>>(sfr, zn);
  }
  // phase 0b: convert the ten 128x128 weights to f16 (pool order = W_* enums)
  {
    WPtrs wp;
    wp.w[W_FRW1] = (const float*)d_in[4];
    wp.w[W_FRW2] = (const float*)d_in[8];
    wp.w[W_FRT1] = (const float*)d_in[6];
    wp.w[W_FRT2] = (const float*)d_in[10];
    wp.w[W_BEW1] = (const float*)d_in[12];
    wp.w[W_BEW2] = (const float*)d_in[16];
    wp.w[W_BET1] = (const float*)d_in[14];
    wp.w[W_BET2] = (const float*)d_in[18];
    wp.w[W_BAL1] = (const float*)d_in[20];
    wp.w[W_SELF] = (const float*)d_in[24];
    lasage_wcvt_kernel<<<(10 * 16384) / 256, 256, 0, stream>>>(wp, wpool);
  }
  // phase 1: label-masked segment sums
  {
    long long threads = (long long)E * 32;
    lasage_scatter_kernel<<<(unsigned)((threads + 255) / 256), 256, 0, stream>>>(
        feat, src, dst, labels, sfr, sbe, sunk, E);
  }
  // phase 2: fused WMMA dense pipeline
  DenseArgs a;
  a.feat  = feat; a.sfr = sfr; a.sbe = sbe; a.sunk = sunk;
  a.wpool = wpool;
  a.frb1  = (const float*)d_in[5];  a.frT1b = (const float*)d_in[7];
  a.frb2  = (const float*)d_in[9];  a.frT2b = (const float*)d_in[11];
  a.beb1  = (const float*)d_in[13]; a.beT1b = (const float*)d_in[15];
  a.beb2  = (const float*)d_in[17]; a.beT2b = (const float*)d_in[19];
  a.balb1 = (const float*)d_in[21]; a.balW2 = (const float*)d_in[22];
  a.balb2 = (const float*)d_in[23]; a.selfb = (const float*)d_in[25];
  a.out   = (float*)d_out;

  lasage_dense_kernel<<<(N + 15) / 16, 256, 0, stream>>>(a, N);
}
